// GraphConvolutionChebyshev_90409061581169
// MI455X (gfx1250) — matrix operand order repaired and checked by reading, not compiled, not run on hardware
//
#include <hip/hip_runtime.h>

#define BATCH 32
#define NN 2048
#define FEAT 64
#define TILE_M 128
#define TILE_K 32

typedef __attribute__((ext_vector_type(16))) __bf16 v16bf;
typedef __attribute__((ext_vector_type(8)))  float  v8f;

struct alignas(16) U4 { unsigned int x, y, z, w; };
struct alignas(16) F4 { float x, y, z, w; };

union Frag { U4 q[2]; v16bf v; };

__device__ __forceinline__ unsigned short f2bf(float f) {
  unsigned int u = __float_as_uint(f);
  u += 0x7fffu + ((u >> 16) & 1u);            // round-to-nearest-even
  return (unsigned short)(u >> 16);
}

// Pack two f32 -> packed bf16x2 with RNE; v_perm_b32 grabs the two high halves.
__device__ __forceinline__ unsigned int pack2(float lo, float hi) {
  unsigned int ul = __float_as_uint(lo);
  unsigned int uh = __float_as_uint(hi);
  ul += 0x7fffu + ((ul >> 16) & 1u);
  uh += 0x7fffu + ((uh >> 16) & 1u);
  return __builtin_amdgcn_perm(uh, ul, 0x07060302u);  // {uh[31:16], ul[31:16]}
}

// Convert + store one staged tile pair into LDS buffers.
// s_a: [TILE_M][TILE_K] bf16 (A layout), s_b: [FEAT][TILE_K] bf16 (B^T layout)
__device__ __forceinline__ void stage_store(
    unsigned short* s_a, unsigned short* s_b,
    int arow, int acol, int krow, int ccol,
    const F4* ra, const F4* rbv)
{
  union { unsigned int u[8]; U4 q[2]; } pk;
  pk.u[0] = pack2(ra[0].x, ra[0].y);
  pk.u[1] = pack2(ra[0].z, ra[0].w);
  pk.u[2] = pack2(ra[1].x, ra[1].y);
  pk.u[3] = pack2(ra[1].z, ra[1].w);
  pk.u[4] = pack2(ra[2].x, ra[2].y);
  pk.u[5] = pack2(ra[2].z, ra[2].w);
  pk.u[6] = pack2(ra[3].x, ra[3].y);
  pk.u[7] = pack2(ra[3].z, ra[3].w);
  *(U4*)&s_a[arow * TILE_K + acol]     = pk.q[0];
  *(U4*)&s_a[arow * TILE_K + acol + 8] = pk.q[1];
  float v[8] = {rbv[0].x, rbv[0].y, rbv[0].z, rbv[0].w,
                rbv[1].x, rbv[1].y, rbv[1].z, rbv[1].w};
  #pragma unroll
  for (int j = 0; j < 8; ++j)
    s_b[(ccol + j) * TILE_K + krow] = f2bf(v[j]);     // transposed scatter
}

// One pipelined step: global-load tile knext -> regs, WMMA from (ca,cb),
// convert+store regs into (na,nb), barrier.
__device__ __forceinline__ void gemm_step(
    const unsigned short* ca, const unsigned short* cb,
    unsigned short* na, unsigned short* nb,
    const float* gA, const float* gB, int knext,
    int arow, int acol, int krow, int ccol,
    int wr, int idx, int half, v8f* acc)
{
  F4 ra[4], rbv[2];
  const bool more = (knext < NN);                     // uniform branch
  if (more) {
    const F4* pa = (const F4*)(gA + knext);
    ra[0] = pa[0]; ra[1] = pa[1]; ra[2] = pa[2]; ra[3] = pa[3];
    const F4* pb = (const F4*)(gB + (size_t)knext * FEAT);
    rbv[0] = pb[0]; rbv[1] = pb[1];
  }
  Frag fa, fb;
  // A (16x32 bf16): VGPR0-3 K={0..7}+8*half, VGPR4-7 K={16..23}+8*half
  fa.q[0] = *(const U4*)&ca[(wr + idx) * TILE_K + half * 8];
  fa.q[1] = *(const U4*)&ca[(wr + idx) * TILE_K + 16 + half * 8];
  #pragma unroll
  for (int t = 0; t < 4; ++t) {
    // B (32x16 bf16): lane=col, VGPR0-3 K={0..7}+8*half, VGPR4-7 K={16..23}+8*half
    fb.q[0] = *(const U4*)&cb[(t * 16 + idx) * TILE_K + half * 8];
    fb.q[1] = *(const U4*)&cb[(t * 16 + idx) * TILE_K + 16 + half * 8];
    acc[t] = __builtin_amdgcn_wmma_f32_16x16x32_bf16(
        false, fa.v, false, fb.v, (short)0, acc[t], false, false);
  }
  if (more) stage_store(na, nb, arow, acol, krow, ccol, ra, rbv);
  __syncthreads();
}

// C(128x64) += adj(128 x NN slice) @ bsrc(NN x 64), double-buffered.
__device__ __forceinline__ void gemm_core(
    const float* __restrict__ adj, const float* __restrict__ bsrc,
    int b, int tileM0,
    unsigned short* s_a0, unsigned short* s_b0,
    unsigned short* s_a1, unsigned short* s_b1, v8f* acc)
{
  const int tid  = threadIdx.x;
  const int lane = tid & 31;
  const int wave = tid >> 5;
  const int idx  = lane & 15;
  const int half = lane >> 4;
  const int wr   = wave * 16;

  // adj staging: 2 threads per row, 16 f32 each
  const int arow = tid >> 1;
  const int acol = (tid & 1) * 16;
  const float* gA = adj + ((size_t)b * NN + tileM0 + arow) * NN + acol;

  // B staging: 8 threads per k-row, 8 f32 each, written transposed
  const int krow = tid >> 3;
  const int ccol = (tid & 7) * 8;
  const float* gB = bsrc + ((size_t)b * NN + krow) * FEAT + ccol;

  // Prologue: stage k=0 into buffer 0
  {
    F4 ra[4], rbv[2];
    const F4* pa = (const F4*)gA;
    ra[0] = pa[0]; ra[1] = pa[1]; ra[2] = pa[2]; ra[3] = pa[3];
    const F4* pb = (const F4*)gB;
    rbv[0] = pb[0]; rbv[1] = pb[1];
    stage_store(s_a0, s_b0, arow, acol, krow, ccol, ra, rbv);
  }
  __syncthreads();

  #pragma unroll 1
  for (int k0 = 0; k0 < NN; k0 += 2 * TILE_K) {
    gemm_step(s_a0, s_b0, s_a1, s_b1, gA, gB, k0 + TILE_K,
              arow, acol, krow, ccol, wr, idx, half, acc);
    gemm_step(s_a1, s_b1, s_a0, s_b0, gA, gB, k0 + 2 * TILE_K,
              arow, acol, krow, ccol, wr, idx, half, acc);
  }
}

// Kernel 1: T1 = adj @ x  (f32 result to workspace)
__global__ __launch_bounds__(256) void cheb_prop(
    const float* __restrict__ adj, const float* __restrict__ x,
    float* __restrict__ t1)
{
  __shared__ alignas(16) unsigned short s_a0[TILE_M * TILE_K];
  __shared__ alignas(16) unsigned short s_a1[TILE_M * TILE_K];
  __shared__ alignas(16) unsigned short s_b0[FEAT * TILE_K];
  __shared__ alignas(16) unsigned short s_b1[FEAT * TILE_K];
  const int b = blockIdx.y;
  const int tileM0 = blockIdx.x * TILE_M;

  v8f acc[4] = {};
  gemm_core(adj, x, b, tileM0, s_a0, s_b0, s_a1, s_b1, acc);

  const int lane = threadIdx.x & 31;
  const int wave = threadIdx.x >> 5;
  const int idx = lane & 15, half = lane >> 4;
  const size_t rb = (size_t)b * NN + tileM0 + wave * 16;
  #pragma unroll
  for (int t = 0; t < 4; ++t)
    #pragma unroll
    for (int r = 0; r < 8; ++r)
      t1[(rb + r + half * 8) * FEAT + t * 16 + idx] = acc[t][r];
}

// Kernel 2: S = adj @ T1; combined = (K0-K2)*x + K1*T1 + 2*K2*S;
//           out = combined @ weight + bias  (second-level WMMA epilogue)
__global__ __launch_bounds__(256) void cheb_out(
    const float* __restrict__ adj, const float* __restrict__ x,
    const float* __restrict__ t1, const float* __restrict__ weight,
    const float* __restrict__ Kp, const float* __restrict__ bias,
    float* __restrict__ out)
{
  __shared__ alignas(16) unsigned short s_a0[TILE_M * TILE_K];
  __shared__ alignas(16) unsigned short s_a1[TILE_M * TILE_K];
  __shared__ alignas(16) unsigned short s_b0[FEAT * TILE_K];
  __shared__ alignas(16) unsigned short s_b1[FEAT * TILE_K];
  __shared__ alignas(16) unsigned short s_w[FEAT * FEAT];    // weight^T bf16
  __shared__ alignas(16) unsigned short s_c[8 * 16 * FEAT];  // per-wave combined

  const int tid = threadIdx.x;
  const int b = blockIdx.y;
  const int tileM0 = blockIdx.x * TILE_M;

  // Stage s_w[o][f] = bf16(weight[f][o]) once (covered by prologue barrier)
  {
    const int f  = tid >> 2;
    const int o0 = (tid & 3) * 16;
    const F4* p = (const F4*)(weight + f * FEAT + o0);
    F4 w0 = p[0], w1 = p[1], w2 = p[2], w3 = p[3];
    float wv[16] = {w0.x, w0.y, w0.z, w0.w, w1.x, w1.y, w1.z, w1.w,
                    w2.x, w2.y, w2.z, w2.w, w3.x, w3.y, w3.z, w3.w};
    #pragma unroll
    for (int j = 0; j < 16; ++j)
      s_w[(o0 + j) * FEAT + f] = f2bf(wv[j]);
  }

  v8f acc[4] = {};
  gemm_core(adj, t1, b, tileM0, s_a0, s_b0, s_a1, s_b1, acc);

  const int lane = tid & 31;
  const int wave = tid >> 5;
  const int idx = lane & 15, half = lane >> 4;
  const size_t rb = (size_t)b * NN + tileM0 + wave * 16;

  const float K0 = Kp[0], K1 = Kp[1], K2 = Kp[2];
  const float c0 = K0 - K2, c1 = K1, c2 = 2.0f * K2;

  // combined tile -> LDS bf16, per-wave region (wave-local LDS is in-order)
  #pragma unroll
  for (int t = 0; t < 4; ++t) {
    #pragma unroll
    for (int r = 0; r < 8; ++r) {
      const int m = r + half * 8;
      const int n = t * 16 + idx;
      const size_t gi = (rb + m) * FEAT + n;
      const float cv = c0 * x[gi] + c1 * t1[gi] + c2 * acc[t][r];
      s_c[(wave * 16 + m) * FEAT + n] = f2bf(cv);
    }
  }

  // out tile accumulators start at bias[n] (column-constant in C/D layout)
  v8f oc[4];
  #pragma unroll
  for (int t = 0; t < 4; ++t) {
    const float bv = bias[t * 16 + idx];
    #pragma unroll
    for (int r = 0; r < 8; ++r) oc[t][r] = bv;
  }

  Frag fa, fb;
  #pragma unroll
  for (int kk = 0; kk < 2; ++kk) {   // k = 0..31, 32..63 of FEAT
    const int cb = (wave * 16 + idx) * FEAT + kk * 32;
    fa.q[0] = *(const U4*)&s_c[cb + half * 8];
    fa.q[1] = *(const U4*)&s_c[cb + 16 + half * 8];
    #pragma unroll
    for (int t = 0; t < 4; ++t) {
      const int wb = (t * 16 + idx) * FEAT + kk * 32;
      fb.q[0] = *(const U4*)&s_w[wb + half * 8];
      fb.q[1] = *(const U4*)&s_w[wb + 16 + half * 8];
      oc[t] = __builtin_amdgcn_wmma_f32_16x16x32_bf16(
          false, fa.v, false, fb.v, (short)0, oc[t], false, false);
    }
  }

  #pragma unroll
  for (int t = 0; t < 4; ++t)
    #pragma unroll
    for (int r = 0; r < 8; ++r)
      out[(rb + r + half * 8) * FEAT + t * 16 + idx] = oc[t][r];
}

extern "C" void kernel_launch(void* const* d_in, const int* in_sizes, int n_in,
                              void* d_out, int out_size, void* d_ws, size_t ws_size,
                              hipStream_t stream) {
  const float* x      = (const float*)d_in[0];   // [32,2048,64]
  const float* adj    = (const float*)d_in[1];   // [32,2048,2048]
  const float* weight = (const float*)d_in[2];   // [64,64]
  const float* Kp     = (const float*)d_in[3];   // [3,1]
  const float* bias   = (const float*)d_in[4];   // [64]
  float* t1 = (float*)d_ws;                      // needs 32*2048*64*4 = 16 MiB

  dim3 grid(NN / TILE_M, BATCH);                 // (16, 32), 256 thr = 8 waves
  cheb_prop<<<grid, 256, 0, stream>>>(adj, x, t1);
  cheb_out<<<grid, 256, 0, stream>>>(adj, x, t1, weight, Kp, bias, (float*)d_out);
}
